// Hahow_Model_62294205661348
// MI455X (gfx1250) — matrix operand order repaired and checked by reading, not compiled, not run on hardware
//
#include <hip/hip_runtime.h>

// ---------------------------------------------------------------------------
// Types for CDNA5 WMMA
// ---------------------------------------------------------------------------
typedef __attribute__((ext_vector_type(16))) __bf16          v16bf;
typedef __attribute__((ext_vector_type(8)))  float           v8f;
typedef __attribute__((ext_vector_type(8)))  unsigned short  ushort8;

union FragU {
    ushort8 h[2];
    v16bf   v;
};

__device__ __forceinline__ unsigned short f2bf(float f) {
    unsigned int u = __builtin_bit_cast(unsigned int, f);
    unsigned int r = (u + 0x7FFFu + ((u >> 16) & 1u)) >> 16;  // round-nearest-even
    return (unsigned short)r;
}

#define K_SMALLEST 45
#define FILL_VAL   0.05f
#define BN_EPS     1e-5f

// ---- CDNA5 async global->LDS path (guarded: falls back to reg staging) ----
#if __has_builtin(__builtin_amdgcn_global_load_async_to_lds_b128)
#define USE_ASYNC_LDS 1
#else
#define USE_ASYNC_LDS 0
#endif

#if USE_ASYNC_LDS
// Builtin signature (from hipcc diagnostic): param0 is AS(1) pointer to a
// 16-byte int vector; by symmetry param1 is the AS(3) LDS pointer, then
// immediate offset + cpol.
typedef int v4i __attribute__((__vector_size__(16)));
typedef __attribute__((address_space(1))) v4i* gas_b128;
typedef __attribute__((address_space(3))) v4i* las_b128;

__device__ __forceinline__ void async_cp16(const unsigned short* g, unsigned short* l) {
    __builtin_amdgcn_global_load_async_to_lds_b128(
        (gas_b128)(void*)g, (las_b128)(void*)l, 0, 0);
}
#if __has_builtin(__builtin_amdgcn_s_wait_asynccnt)
#define WAIT_ASYNC0() __builtin_amdgcn_s_wait_asynccnt(0)
#else
#define WAIT_ASYNC0() asm volatile("s_wait_asynccnt 0" ::: "memory")
#endif
#endif

// ---------------------------------------------------------------------------
// Prep: fold BN (inference) into per-output-row weight scale + fused bias.
//   bn(y) = y*s + t,  s = gamma*rsqrt(var+eps),  t = beta - mean*s
// Same BN params are used for both layers in the reference.
// ---------------------------------------------------------------------------
__global__ void bn_prep(const float* __restrict__ gamma, const float* __restrict__ beta,
                        const float* __restrict__ mean,  const float* __restrict__ var,
                        const float* __restrict__ b1,    const float* __restrict__ b2,
                        float* __restrict__ s, float* __restrict__ bias1,
                        float* __restrict__ bias2, int H) {
    int i = blockIdx.x * blockDim.x + threadIdx.x;
    if (i < H) {
        float sc = gamma[i] * rsqrtf(var[i] + BN_EPS);
        float t  = beta[i] - mean[i] * sc;
        s[i]     = sc;
        bias1[i] = b1[i] * sc + t;
        bias2[i] = b2[i] * sc + t;
    }
}

// f32 -> bf16 with optional per-row scale (row = idx / K). s==nullptr -> scale 1.
__global__ void wconv(const float* __restrict__ W, const float* __restrict__ s,
                      unsigned short* __restrict__ out, int N, int K) {
    long long idx   = (long long)blockIdx.x * blockDim.x + threadIdx.x;
    long long total = (long long)N * K;
    if (idx < total) {
        int   n  = (int)(idx / K);
        float sc = s ? s[n] : 1.0f;
        out[idx] = f2bf(W[idx] * sc);
    }
}

// ---------------------------------------------------------------------------
// bf16 WMMA GEMM:  out[M,N] = A[M,K] * Bw[N,K]^T (+bias) (+relu)
//   EPI 0: bias + relu -> bf16 out
//   EPI 1: bias        -> f32 out   (logits)
//   EPI 2: raw         -> f32 out   (rt)
// Block = 256 threads = 8 waves (4M x 2N). Block tile 128x128, wave tile 32x64
// (2x4 WMMA accumulators), K-step 32, double-buffered LDS (32 KB) fed by
// async global->LDS copies (ASYNCcnt) when available.
// ---------------------------------------------------------------------------
template <int EPI>
__global__ void __launch_bounds__(256)
gemm_bf16(const unsigned short* __restrict__ A, const unsigned short* __restrict__ Bw,
          const float* __restrict__ bias, void* __restrict__ outv,
          int M, int N, int K) {
    __shared__ __align__(16) unsigned short As[2][128][32];
    __shared__ __align__(16) unsigned short Bs[2][128][32];

    const int tid = threadIdx.x;
    const int bm  = blockIdx.x * 128;
    const int bn  = blockIdx.y * 128;

    // ---- loader mapping: A and B tiles are each 128x32 ushort
    //      = 512 x 16B chunks = 2 chunks/thread/tile
    const int r0 = tid >> 2;             // rows 0..63
    const int r1 = (tid + 256) >> 2;     // rows 64..127
    const int co = (tid & 3) * 8;        // ushort offset in row (0,8,16,24)
    const unsigned short* Ap0 = A  + (size_t)(bm + r0) * K + co;
    const unsigned short* Ap1 = A  + (size_t)(bm + r1) * K + co;
    const unsigned short* Bp0 = Bw + (size_t)(bn + r0) * K + co;
    const unsigned short* Bp1 = Bw + (size_t)(bn + r1) * K + co;

    // ---- wave mapping: 8 waves = 4(M) x 2(N); wave tile 32(M) x 64(N)
    const int lane = tid & 31;
    const int half = lane >> 4;
    const int lr   = lane & 15;
    const int wave = tid >> 5;
    const int wm   = (wave & 3) * 32;
    const int wn   = (wave >> 2) * 64;

    v8f acc[2][4] = {};
    const int nk = K >> 5;

#if USE_ASYNC_LDS
    // issue tile 0
    async_cp16(Ap0, &As[0][r0][co]);
    async_cp16(Ap1, &As[0][r1][co]);
    async_cp16(Bp0, &Bs[0][r0][co]);
    async_cp16(Bp1, &Bs[0][r1][co]);
#else
    float4 pa0 = *(const float4*)(Ap0);
    float4 pa1 = *(const float4*)(Ap1);
    float4 pb0 = *(const float4*)(Bp0);
    float4 pb1 = *(const float4*)(Bp1);
    *(float4*)&As[0][r0][co] = pa0;
    *(float4*)&As[0][r1][co] = pa1;
    *(float4*)&Bs[0][r0][co] = pb0;
    *(float4*)&Bs[0][r1][co] = pb1;
#endif

    for (int ks = 0; ks < nk; ++ks) {
        const int buf = ks & 1;

#if USE_ASYNC_LDS
        WAIT_ASYNC0();            // this wave's LDS writes for buf complete
        __syncthreads();          // everyone done writing buf / reading buf^1
        if (ks + 1 < nk) {        // prefetch next tile into buf^1 (overlaps compute)
            const int k = (ks + 1) << 5;
            const int nb = buf ^ 1;
            async_cp16(Ap0 + k, &As[nb][r0][co]);
            async_cp16(Ap1 + k, &As[nb][r1][co]);
            async_cp16(Bp0 + k, &Bs[nb][r0][co]);
            async_cp16(Bp1 + k, &Bs[nb][r1][co]);
            if (ks + 2 < nk) {
                const int k2 = (ks + 2) << 5;
                __builtin_prefetch(Ap0 + k2, 0, 0);   // global_prefetch_b8 (L2)
                __builtin_prefetch(Bp0 + k2, 0, 0);
            }
        }
#else
        __syncthreads();
        if (ks + 1 < nk) {
            const int k = (ks + 1) << 5;
            pa0 = *(const float4*)(Ap0 + k);
            pa1 = *(const float4*)(Ap1 + k);
            pb0 = *(const float4*)(Bp0 + k);
            pb1 = *(const float4*)(Bp1 + k);
            if (ks + 2 < nk) {
                const int k2 = (ks + 2) << 5;
                __builtin_prefetch(Ap0 + k2, 0, 0);
                __builtin_prefetch(Bp0 + k2, 0, 0);
            }
        }
#endif

        // ---- fragments per ISA 7.12.2 layouts
        FragU af[2], bfg[4];
#pragma unroll
        for (int mi = 0; mi < 2; ++mi) {
            const unsigned short* rp = &As[buf][wm + mi * 16 + lr][0];
            af[mi].h[0] = *(const ushort8*)(rp + half * 8);       // K 0-7 / 8-15
            af[mi].h[1] = *(const ushort8*)(rp + 16 + half * 8);  // K 16-23 / 24-31
        }
#pragma unroll
        for (int ni = 0; ni < 4; ++ni) {
            const unsigned short* rp = &Bs[buf][wn + ni * 16 + lr][0];
            bfg[ni].h[0] = *(const ushort8*)(rp + half * 8);      // K 0-7 / 8-15
            bfg[ni].h[1] = *(const ushort8*)(rp + 16 + half * 8); // K 16-23 / 24-31
        }

#pragma unroll
        for (int mi = 0; mi < 2; ++mi)
#pragma unroll
            for (int ni = 0; ni < 4; ++ni)
                acc[mi][ni] = __builtin_amdgcn_wmma_f32_16x16x32_bf16(
                    false, af[mi].v, false, bfg[ni].v,
                    (short)0, acc[mi][ni], false, false);

#if !USE_ASYNC_LDS
        if (ks + 1 < nk) {
            const int nb = (ks + 1) & 1;
            *(float4*)&As[nb][r0][co] = pa0;
            *(float4*)&As[nb][r1][co] = pa1;
            *(float4*)&Bs[nb][r0][co] = pb0;
            *(float4*)&Bs[nb][r1][co] = pb1;
        }
#endif
    }

    // ---- epilogue (C/D layout: VGPR i -> M=i (lanes 0-15) / M=8+i (lanes 16-31))
    unsigned short* outU = (unsigned short*)outv;
    float*          outF = (float*)outv;
#pragma unroll
    for (int ni = 0; ni < 4; ++ni) {
        const int gn = bn + wn + ni * 16 + lr;
        float bv = 0.0f;
        if constexpr (EPI != 2) bv = bias[gn];
#pragma unroll
        for (int mi = 0; mi < 2; ++mi) {
#pragma unroll
            for (int i = 0; i < 8; ++i) {
                const int gm  = bm + wm + mi * 16 + i + half * 8;
                float val = acc[mi][ni][i] + bv;
                if constexpr (EPI == 0) {
                    val = fmaxf(val, 0.0f);
                    outU[(size_t)gm * N + gn] = f2bf(val);
                } else {
                    outF[(size_t)gm * N + gn] = val;
                }
            }
        }
    }
}

// ---------------------------------------------------------------------------
// Per-row 45-smallest -> scatter 0.05. rank = #{w<v} + #{w==v && j<i} matches
// top_k(-logits) tie-break exactly. 128 threads per row (C=128).
// ---------------------------------------------------------------------------
__global__ void __launch_bounds__(128)
topk_select(const float* __restrict__ logits, unsigned short* __restrict__ pb) {
    __shared__ float row[128];
    const int r = blockIdx.x;
    const int t = threadIdx.x;
    const float v = logits[(size_t)r * 128 + t];
    row[t] = v;
    __syncthreads();
    int rank = 0;
#pragma unroll 8
    for (int j = 0; j < 128; ++j) {
        const float w = row[j];
        rank += (w < v) || (w == v && j < t);
    }
    const float p = (rank < K_SMALLEST) ? FILL_VAL : v;
    pb[(size_t)r * 128 + t] = f2bf(p);
}

// ---------------------------------------------------------------------------
// Launch
// ---------------------------------------------------------------------------
extern "C" void kernel_launch(void* const* d_in, const int* in_sizes, int n_in,
                              void* d_out, int out_size, void* d_ws, size_t ws_size,
                              hipStream_t stream) {
    (void)in_sizes; (void)n_in; (void)out_size; (void)ws_size;
    const int B = 16384, F = 512, H = 1024, C = 128, NC = 4096;

    const float* x     = (const float*)d_in[0];
    const float* W1    = (const float*)d_in[1];
    const float* b1    = (const float*)d_in[2];
    const float* W2    = (const float*)d_in[3];
    const float* b2    = (const float*)d_in[4];
    const float* W4    = (const float*)d_in[5];
    const float* b4    = (const float*)d_in[6];
    const float* gamma = (const float*)d_in[7];
    const float* beta  = (const float*)d_in[8];
    const float* rmean = (const float*)d_in[9];
    const float* rvar  = (const float*)d_in[10];
    const float* tc    = (const float*)d_in[11];

    // workspace layout (256B aligned)
    char*  w   = (char*)d_ws;
    size_t off = 0;
    auto alloc = [&](size_t bytes) -> void* {
        void* p = w + off;
        off = (off + bytes + 255) & ~(size_t)255;
        return p;
    };
    float* s      = (float*)alloc((size_t)H * 4);
    float* bias1  = (float*)alloc((size_t)H * 4);
    float* bias2  = (float*)alloc((size_t)H * 4);
    unsigned short* W1b = (unsigned short*)alloc((size_t)H * F * 2);
    unsigned short* W2b = (unsigned short*)alloc((size_t)H * H * 2);
    unsigned short* W4b = (unsigned short*)alloc((size_t)C * H * 2);
    unsigned short* TCb = (unsigned short*)alloc((size_t)NC * C * 2);
    unsigned short* Xb  = (unsigned short*)alloc((size_t)B * F * 2);
    unsigned short* H1b = (unsigned short*)alloc((size_t)B * H * 2);
    unsigned short* H2b = (unsigned short*)alloc((size_t)B * H * 2);
    unsigned short* Pb  = (unsigned short*)alloc((size_t)B * C * 2);

    // ---- prep
    bn_prep<<<(H + 255) / 256, 256, 0, stream>>>(gamma, beta, rmean, rvar, b1, b2,
                                                 s, bias1, bias2, H);
    auto cgrid = [](long long total) { return (unsigned)((total + 255) / 256); };
    wconv<<<cgrid((long long)H * F),  256, 0, stream>>>(W1, s,       W1b, H,  F);
    wconv<<<cgrid((long long)H * H),  256, 0, stream>>>(W2, s,       W2b, H,  H);
    wconv<<<cgrid((long long)C * H),  256, 0, stream>>>(W4, nullptr, W4b, C,  H);
    wconv<<<cgrid((long long)NC * C), 256, 0, stream>>>(tc, nullptr, TCb, NC, C);
    wconv<<<cgrid((long long)B * F),  256, 0, stream>>>(x,  nullptr, Xb,  B,  F);

    float* logits = (float*)d_out;
    float* rt     = (float*)d_out + (size_t)B * C;

    dim3 blk(256);
    // layer 1: h1 = relu(bn(x @ W1^T + b1))
    gemm_bf16<0><<<dim3(B / 128, H / 128), blk, 0, stream>>>(Xb,  W1b, bias1, H1b, B, H, F);
    // layer 2: h2 = relu(bn(h1 @ W2^T + b2))
    gemm_bf16<0><<<dim3(B / 128, H / 128), blk, 0, stream>>>(H1b, W2b, bias2, H2b, B, H, H);
    // logits = h2 @ W4^T + b4  (f32, straight into d_out)
    gemm_bf16<1><<<dim3(B / 128, C / 128), blk, 0, stream>>>(H2b, W4b, b4, logits, B, C, H);
    // top-45-smallest -> 0.05 scatter -> bf16 p
    topk_select<<<B, 128, 0, stream>>>(logits, Pb);
    // rt = p @ topic_course^T
    gemm_bf16<2><<<dim3(B / 128, NC / 128), blk, 0, stream>>>(Pb, TCb, nullptr, rt, B, NC, C);
}
